// DCT_46119358825006
// MI455X (gfx1250) — matrix operand order repaired and checked by reading, not compiled, not run on hardware
//
#include <hip/hip_runtime.h>

typedef __attribute__((ext_vector_type(16))) _Float16 v16h;
typedef __attribute__((ext_vector_type(8)))  float    v8f;

#define PI_F 3.14159265358979323846f

// Problem geometry (fixed by the reference: 1x4096x4096, BLOCK=8)
// GEMM per j:  Y_j[k, r] = sum_w Dw[k,w] * floor(255*x[r, 8w+j])
//   M = 512 (k), K = 512 (w), N = 4096 (r)
// Epilogue:    out[8h+m, 8k+j] = mask(m,j) * sum_i Db[m,i] * Y_j[k, 8h+i]

constexpr int MT_TOT = 32;   // 512 / 16 M-tiles
constexpr int KT_TOT = 16;   // 512 / 32 K-steps
constexpr int NT_TOT = 256;  // 4096 / 16 N-tiles
constexpr int A_HALVES = MT_TOT * KT_TOT * 512; // 262144 halves (512 KB)

// Per-kt strides in halves (used as immediate offsets in the main loop)
constexpr int A_KT_STRIDE = 512;          // (mt*16 + kt)*512 : +512 per kt
constexpr int B_KT_STRIDE = 256 * 512;    // ((j*16+kt)*256 + nt)*512 : +131072 per kt

// ---------------------------------------------------------------------------
// Generate Dw (512-pt DCT-II matrix) directly in WMMA A-fragment order.
// A fragment (16x32 f16): lane L -> row M = L%16; half t -> K_local:
//   t<8 : K = (L>=16 ? 8:0)  + t
//   t>=8: K = (L>=16 ? 24:16) + (t-8)
// ---------------------------------------------------------------------------
__global__ void pack_dw_kernel(_Float16* __restrict__ ap) {
    int idx  = blockIdx.x * 256 + threadIdx.x;      // < 262144
    int t    = idx & 15;
    int lane = (idx >> 4) & 31;
    int kt   = (idx >> 9) & 15;
    int mt   = idx >> 13;
    int m    = mt * 16 + (lane & 15);
    int hi   = lane >> 4;
    int kl   = (t < 8) ? ((hi ? 8 : 0) + t) : ((hi ? 24 : 16) + (t - 8));
    int w    = kt * 32 + kl;
    float scale = (m == 0) ? 0.044194173824159223f   // sqrt(1/512)
                           : 0.0625f;                // sqrt(2/512)
    float v = cosf(PI_F * (2.0f * (float)w + 1.0f) * (float)m * (1.0f / 1024.0f)) * scale;
    ap[idx] = (_Float16)v;
}

// ---------------------------------------------------------------------------
// Pack x (f32, quantized by floor(255*x)) into WMMA B-fragment order for all
// 8 j-GEMMs. Coalesced reads of x; 2-byte scattered writes absorbed by L2.
// B fragment (32x16 f16): lane L -> col N = L%16; t<8: K = (L>=16 ? 8:0) + t; t>=8: K = (L>=16 ? 24:16) + (t-8).
// bp layout (halves): [j][kt][nt][lane][t]
// ---------------------------------------------------------------------------
__global__ void pack_x_kernel(const float* __restrict__ x, _Float16* __restrict__ bp) {
    int idx = blockIdx.x * 256 + threadIdx.x;       // < 4096*4096
    int r   = idx >> 12;                            // row of x  (= N index)
    int c   = idx & 4095;                           // col of x
    int j   = c & 7;
    int w   = c >> 3;                               // K index
    int kt  = w >> 5;
    int kk  = w & 31;
    int lane = (r & 15) + (((kk >> 3) & 1) << 4);
    int t    = (kk & 7) + ((kk >> 4) << 3);
    int nt   = r >> 4;
    int dst  = ((((j * 16 + kt) * 256 + nt) * 32) + lane) * 16 + t;
    bp[dst] = (_Float16)floorf(x[idx] * 255.0f);
}

// ---------------------------------------------------------------------------
// Fragment load / MMA helpers for the double-buffered pipeline.
// aB[mi], bB[ni] are per-wave base pointers; kt selects an immediate offset.
// ---------------------------------------------------------------------------
__device__ __forceinline__ void load_frags(const _Float16* const aB[4],
                                           const _Float16* const bB[2],
                                           int kt, v16h a[4], v16h b[2]) {
    #pragma unroll
    for (int mi = 0; mi < 4; ++mi)
        a[mi] = *(const v16h*)(aB[mi] + kt * A_KT_STRIDE);
    #pragma unroll
    for (int ni = 0; ni < 2; ++ni)
        b[ni] = *(const v16h*)(bB[ni] + kt * B_KT_STRIDE);
}

__device__ __forceinline__ void mma_stage(v8f c[4][2], const v16h a[4], const v16h b[2]) {
    #pragma unroll
    for (int mi = 0; mi < 4; ++mi)
        #pragma unroll
        for (int ni = 0; ni < 2; ++ni)
            c[mi][ni] = __builtin_amdgcn_wmma_f32_16x16x32_f16(
                false, a[mi], false, b[ni],
                (short)0, c[mi][ni], false, false);
}

// ---------------------------------------------------------------------------
// Main GEMM + fused Db epilogue.
// grid = (128 N-panels of 32 cols, 8 j), block = 256 (8 waves).
// Wave wv owns M rows [wv*64, wv*64+64): 4 (M) x 2 (N) wmma tiles.
// ---------------------------------------------------------------------------
__global__ __launch_bounds__(256) void dct_gemm_kernel(
        const _Float16* __restrict__ ap,
        const _Float16* __restrict__ bp,
        float* __restrict__ out) {
    __shared__ float ylds[8 * 64 * 32];             // 64 KB: Y panel [512 x 32]

    const int tid   = threadIdx.x;
    const int lane  = tid & 31;
    const int wv    = tid >> 5;
    const int j     = blockIdx.y;
    const int panel = blockIdx.x;                   // 32-column N panel

    v8f c[4][2] = {};

    if (j >= 2) {                                   // j<2 is fully masked -> skip GEMM
        // Per-fragment base pointers (kt = 0); kt advances via immediate offsets.
        const _Float16* aB[4];
        const _Float16* bB[2];
        #pragma unroll
        for (int mi = 0; mi < 4; ++mi)
            aB[mi] = ap + ((((wv * 4 + mi) * 16) * 32) + lane) * 16;
        #pragma unroll
        for (int ni = 0; ni < 2; ++ni)
            bB[ni] = bp + ((((j * 16) * 256 + panel * 2 + ni) * 32) + lane) * 16;

        // Software-pipelined, double-buffered K loop (16 steps of K=32).
        v16h a0[4], b0[2], a1[4], b1[2];
        load_frags(aB, bB, 0, a0, b0);
        #pragma unroll
        for (int kt = 0; kt < KT_TOT; kt += 2) {
            load_frags(aB, bB, kt + 1, a1, b1);     // prefetch stage kt+1
            mma_stage(c, a0, b0);                   // compute stage kt
            if (kt + 2 < KT_TOT)
                load_frags(aB, bB, kt + 2, a0, b0); // prefetch stage kt+2
            mma_stage(c, a1, b1);                   // compute stage kt+1
        }
    }

    // Stage C fragments to LDS: ylds[k_local(512 via wv)][r_local(32)]
    const int ncol = lane & 15;
    const int mofs = (lane >> 4) * 8;
    #pragma unroll
    for (int mi = 0; mi < 4; ++mi)
        #pragma unroll
        for (int ni = 0; ni < 2; ++ni)
            #pragma unroll
            for (int v = 0; v < 8; ++v)
                ylds[wv * 2048 + (mi * 16 + mofs + v) * 32 + ni * 16 + ncol] = c[mi][ni][v];
    __syncthreads();

    // Epilogue: fold 8 rows (i) through masked Db, scatter to out.
    // Per-thread constants: m = tid%8, h_local = (tid%32)/8, kbase = tid/32.
    const int m     = tid & 7;
    const int hl    = (tid & 31) >> 3;
    const int kbase = tid >> 5;
    float db[8];
    {
        float scale = (m == 0) ? 0.35355339059327373f : 0.5f;  // sqrt(1/8), sqrt(2/8)
        #pragma unroll
        for (int i = 0; i < 8; ++i)
            db[i] = cosf(PI_F * (2.0f * (float)i + 1.0f) * (float)m * (1.0f / 16.0f)) * scale;
    }
    const bool zero = (m < 2) || (j < 2);
    const int hg = panel * 4 + hl;                  // global 8-row block index
    for (int e = 0; e < 64; ++e) {
        int k = e * 8 + kbase;                      // output frequency (col block)
        float s = 0.0f;
        if (!zero) {
            int yb = (k >> 6) * 2048 + (k & 63) * 32 + hl * 8;
            #pragma unroll
            for (int i = 0; i < 8; ++i)
                s += db[i] * ylds[yb + i];
        }
        out[(hg * 8 + m) * 4096 + k * 8 + j] = s;
    }
}

extern "C" void kernel_launch(void* const* d_in, const int* in_sizes, int n_in,
                              void* d_out, int out_size, void* d_ws, size_t ws_size,
                              hipStream_t stream) {
    (void)in_sizes; (void)n_in; (void)out_size; (void)ws_size;
    const float* x   = (const float*)d_in[0];
    float*       out = (float*)d_out;
    _Float16*    ap  = (_Float16*)d_ws;             // 512 KB  packed Dw
    _Float16*    bp  = ap + A_HALVES;               // 32 MB   packed B (all 8 j)

    pack_dw_kernel<<<A_HALVES / 256, 256, 0, stream>>>(ap);
    pack_x_kernel<<<(4096 * 4096) / 256, 256, 0, stream>>>(x, bp);

    dim3 grid(128, 8);
    dct_gemm_kernel<<<grid, 256, 0, stream>>>(ap, bp, out);
}
